// UpGTOv1_22041772163217
// MI455X (gfx1250) — compile-verified, matrix operand order checked by reading, not run on hardware
//
#include <hip/hip_runtime.h>
#include <hip/hip_bf16.h>

// ---------------------------------------------------------------------------
// Fused BN+ReLU -> {ConvTranspose shortcut, ConvTranspose q, k/v proj,
// 8-head flash attention, output projection} for MI455X (gfx1250, wave32).
// All contractions: v_wmma_f32_16x16x32_f16. Operands stored K-minor so every
// A/B build is 2x global/ds b128 loads. Activation tiles staged to LDS with
// global_load_async_to_lds_b128 (ASYNCcnt) and shared across waves.
// ---------------------------------------------------------------------------

typedef __attribute__((ext_vector_type(16))) _Float16 v16h;
typedef __attribute__((ext_vector_type(8)))  _Float16 h8;
typedef __attribute__((ext_vector_type(8)))  float    v8f;

#define BB   4
#define CIN  256
#define HWIN 1024      // 32*32
#define CO   128
#define DK   64
#define DV   64
#define HEADS 8
#define HU   65
#define NQ   4225      // 65*65
#define NK   1024
#define QTILES 265     // ceil(4225/16)

#define WMMA(a, b, c) __builtin_amdgcn_wmma_f32_16x16x32_f16(false, a, false, b, (short)0, c, false, false)
#define SHUF16(l, h) __builtin_shufflevector(l, h, 0,1,2,3,4,5,6,7,8,9,10,11,12,13,14,15)

// low 32 bits of a flat pointer into LDS == wave-relative LDS byte address
__device__ __forceinline__ unsigned lds_addr32(const void* p) {
  return (unsigned)(size_t)p;
}

// Stage `bytes` (multiple of 16, contiguous) from global to LDS, block-wide.
__device__ __forceinline__ void stage_async(const void* gsrc, void* lds, unsigned bytes) {
  for (unsigned off = threadIdx.x * 16u; off < bytes; off += blockDim.x * 16u) {
    unsigned la = lds_addr32((const char*)lds + off);
    const void* ga = (const char*)gsrc + off;
    asm volatile("global_load_async_to_lds_b128 %0, %1, off" :: "v"(la), "v"(ga) : "memory");
  }
  asm volatile("s_wait_asynccnt 0" ::: "memory");
  __syncthreads();
}

// ---------------------------------------------------------------------------
__global__ void zero2_kernel(float* __restrict__ a, long na,
                             float* __restrict__ b, long nb) {
  long i = (long)blockIdx.x * blockDim.x + threadIdx.x;
  if (i < na) a[i] = 0.0f;
  if (i < nb) b[i] = 0.0f;
}

// xr_t[b][sp][c] = relu(bn(x[b][c][sp]))  (K-minor for B operands)
__global__ void bn_relu_pack_kernel(const float* __restrict__ x,
                                    const float* __restrict__ gamma,
                                    const float* __restrict__ beta,
                                    const float* __restrict__ mean,
                                    const float* __restrict__ var,
                                    _Float16* __restrict__ xr_t) {
  int i = blockIdx.x * blockDim.x + threadIdx.x;    // ((b*HWIN)+sp)*CIN + c
  int c  = i & (CIN - 1);
  int sp = (i >> 8) & (HWIN - 1);
  int b  = i >> 18;
  float inv = rsqrtf(var[c] + 1e-5f);
  float v = (x[((b * CIN + c) << 10) + sp] - mean[c]) * (gamma[c] * inv) + beta[c];
  xr_t[i] = (_Float16)(v > 0.0f ? v : 0.0f);
}

// wT[kidx][co][c] = w[c][co][ky][kx]   (K-minor A operand for conv-transpose)
__global__ void pack_wT_kernel(const float* __restrict__ w,
                               _Float16* __restrict__ wT, int Cout) {
  int i = blockIdx.x * blockDim.x + threadIdx.x;    // (kidx*Cout + co)*CIN + c
  if (i >= 9 * Cout * CIN) return;
  int c = i & (CIN - 1);
  int co = (i >> 8) % Cout;
  int kidx = i / (Cout * CIN);
  wT[i] = (_Float16)w[(c * Cout + co) * 9 + kidx];
}

__global__ void cvt_f16_kernel(const float* __restrict__ s,
                               _Float16* __restrict__ d, int n) {
  int i = blockIdx.x * blockDim.x + threadIdx.x;
  if (i < n) d[i] = (_Float16)s[i];
}

// q_t[b][qp][d] = q_acc[b][d][qp] * 1/sqrt(8)   (K-minor, pre-scaled)
__global__ void pack_q_kernel(const float* __restrict__ q_acc,
                              _Float16* __restrict__ q_t) {
  int i = blockIdx.x * blockDim.x + threadIdx.x;    // ((b*NQ)+qp)*64 + d
  if (i >= BB * NQ * DK) return;
  int d = i & 63;
  int qp = (i >> 6) % NQ;
  int b = i / (NQ * DK);
  q_t[i] = (_Float16)(q_acc[(b * DK + d) * NQ + qp] * 0.35355339059327373f);
}

// ---------------------------------------------------------------------------
// ConvTranspose2d(256->Cout, 3x3, s2, VALID) as 9 scattered GEMMs.
// Block = Cout/16 waves; the 8KB activation tile [16 sp][256 c] is staged once
// via async-to-LDS and reused by every (wave=co-tile, kidx) pair.
// jax conv_transpose (no kernel flip) => oy = 2*iy + (2-ky).
__global__ void convt_wmma_kernel(const _Float16* __restrict__ xr_t, // [B][1024][256]
                                  const _Float16* __restrict__ wT,   // [9][Cout][256]
                                  float* __restrict__ dst, int Cout) {
  __shared__ _Float16 sm[16 * CIN];                 // 8 KB
  const int ntile = blockIdx.x, b = blockIdx.z;
  stage_async(xr_t + ((size_t)b * HWIN + ntile * 16) * CIN, sm, 16 * CIN * 2);

  const int lane = threadIdx.x & 31;
  const int wave = threadIdx.x >> 5;                // co-tile
  const int hi = lane >> 4, lo = lane & 15;
  const int nsp = ntile * 16 + lo;
  const int iy = nsp >> 5, ix = nsp & 31;
  const int co_a = wave * 16 + lo;                  // A row (M)

  for (int kidx = 0; kidx < 9; ++kidx) {
    const _Float16* wrow = wT + ((size_t)kidx * Cout + co_a) * CIN;
    __builtin_prefetch(wT + ((size_t)((kidx + 1) % 9) * Cout + co_a) * CIN, 0, 1);
    v8f acc = {};
#pragma unroll
    for (int kc = 0; kc < CIN; kc += 32) {
      h8 a0 = *(const h8*)(wrow + kc + 8 * hi);
      h8 a1 = *(const h8*)(wrow + kc + 16 + 8 * hi);
      h8 b0 = *(const h8*)(sm + lo * CIN + kc + 8 * hi);
      h8 b1 = *(const h8*)(sm + lo * CIN + kc + 16 + 8 * hi);
      v16h av = SHUF16(a0, a1);
      v16h bv = SHUF16(b0, b1);
      acc = WMMA(av, bv, acc);
    }
    const int ky = kidx / 3, kx = kidx % 3;
    const int oy = 2 * iy + (2 - ky), ox = 2 * ix + (2 - kx);
    float* o = dst + (size_t)b * Cout * NQ + (size_t)(oy * HU + ox);
#pragma unroll
    for (int r = 0; r < 8; ++r)
      atomicAdd(o + (size_t)(wave * 16 + r + 8 * hi) * NQ, acc[r]);
  }
}

// ---------------------------------------------------------------------------
// k/v projection: D[d, sp] = sum_c W[d,c] * xr[b,sp,c]; M=64, N=1024, K=256.
// Block = 4 waves sharing the staged activation tile.
// tstore=1 -> K-minor store k_t[b][sp][d]; tstore=0 -> v[b][d][sp].
__global__ void kv_wmma_kernel(const _Float16* __restrict__ xr_t,
                               const _Float16* __restrict__ wh,     // [64][256]
                               _Float16* __restrict__ dst, int tstore) {
  __shared__ _Float16 sm[16 * CIN];
  const int ntile = blockIdx.x, b = blockIdx.z;
  stage_async(xr_t + ((size_t)b * HWIN + ntile * 16) * CIN, sm, 16 * CIN * 2);

  const int lane = threadIdx.x & 31;
  const int wave = threadIdx.x >> 5;                // m-tile
  const int hi = lane >> 4, lo = lane & 15;
  const int nsp = ntile * 16 + lo;
  const _Float16* wrow = wh + (size_t)(wave * 16 + lo) * CIN;

  v8f acc = {};
#pragma unroll
  for (int kc = 0; kc < CIN; kc += 32) {
    h8 a0 = *(const h8*)(wrow + kc + 8 * hi);
    h8 a1 = *(const h8*)(wrow + kc + 16 + 8 * hi);
    h8 b0 = *(const h8*)(sm + lo * CIN + kc + 8 * hi);
    h8 b1 = *(const h8*)(sm + lo * CIN + kc + 16 + 8 * hi);
    acc = WMMA(SHUF16(a0, a1), SHUF16(b0, b1), acc);
  }
  if (tstore) {                                     // k_t[b][sp][64], one b128
    h8 p;
#pragma unroll
    for (int r = 0; r < 8; ++r) p[r] = (_Float16)acc[r];
    *(h8*)(dst + (((size_t)b * HWIN) + nsp) * DK + wave * 16 + 8 * hi) = p;
  } else {                                          // v[b][64][1024]
#pragma unroll
    for (int r = 0; r < 8; ++r)
      dst[((b * DV + wave * 16 + r + 8 * hi) << 10) + nsp] = (_Float16)acc[r];
  }
}

// ---------------------------------------------------------------------------
// Flash attention, one wave per (b, head, 16-query tile). dk=dv=8 (K padded).
__global__ __launch_bounds__(32)
void attn_flash_kernel(const _Float16* __restrict__ q_t,  // [B][4225][64]
                       const _Float16* __restrict__ k_t,  // [B][1024][64]
                       const _Float16* __restrict__ vf,   // [B][64][1024]
                       _Float16* __restrict__ a_t) {      // [B][4225][64]
  __shared__ _Float16 pbuf[16 * 32];
  const int lane = threadIdx.x;
  const int hi = lane >> 4, lo = lane & 15;
  const int qtile = blockIdx.x, head = blockIdx.y, b = blockIdx.z;

  // A operand: q tile [16q x 32K], head-dim slots 0..7 valid (hi==0, halves 0..7)
  v16h qa = {};
  const int qpos0 = qtile * 16 + lo;
  if (hi == 0 && qpos0 < NQ) {
    h8 q8 = *(const h8*)(q_t + ((size_t)b * NQ + qpos0) * DK + head * 8);
#pragma unroll
    for (int i = 0; i < 8; ++i) qa[i] = q8[i];
  }
  const _Float16* vp = vf + (size_t)(b * DV + head * 8) * NK;

  v8f acc = {};
  float m_run[8], l_run[8];
#pragma unroll
  for (int r = 0; r < 8; ++r) { m_run[r] = -3.0e38f; l_run[r] = 0.0f; }

  for (int k0 = 0; k0 < NK; k0 += 32) {
    v8f s[2];
#pragma unroll
    for (int c2 = 0; c2 < 2; ++c2) {
      v16h kb = {};                                  // B[k=d, n=key]
      if (hi == 0) {
        h8 k8 = *(const h8*)(k_t + ((size_t)b * HWIN + k0 + c2 * 16 + lo) * DK + head * 8);
#pragma unroll
        for (int i = 0; i < 8; ++i) kb[i] = k8[i];
      }
      v8f z = {};
      s[c2] = WMMA(qa, kb, z);
    }
    float alpha[8];
#pragma unroll
    for (int r = 0; r < 8; ++r) {                    // row max over 16 N-lanes
      float mx = fmaxf(s[0][r], s[1][r]);
#pragma unroll
      for (int d = 1; d < 16; d <<= 1) mx = fmaxf(mx, __shfl_xor(mx, d, 32));
      float mnew = fmaxf(m_run[r], mx);
      alpha[r] = __expf(m_run[r] - mnew);
      m_run[r] = mnew;
    }
    __syncthreads();
#pragma unroll
    for (int c2 = 0; c2 < 2; ++c2)
#pragma unroll
      for (int r = 0; r < 8; ++r) {
        float p = __expf(s[c2][r] - m_run[r]);
        s[c2][r] = p;
        pbuf[(r + 8 * hi) * 32 + c2 * 16 + lo] = (_Float16)p;  // [m][k]
      }
#pragma unroll
    for (int r = 0; r < 8; ++r) {
      float rs = s[0][r] + s[1][r];
#pragma unroll
      for (int d = 1; d < 16; d <<= 1) rs += __shfl_xor(rs, d, 32);
      l_run[r] = l_run[r] * alpha[r] + rs;
      acc[r] *= alpha[r];
    }
    __syncthreads();
    // A = P (16q x 32key) from LDS (2x ds b128); B = V^T (32key x 16dv)
    h8 p0 = *(const h8*)(pbuf + lo * 32 + 8 * hi);
    h8 p1 = *(const h8*)(pbuf + lo * 32 + 16 + 8 * hi);
    v16h vb = {};
    if (lo < 8) {
      h8 v0 = *(const h8*)(vp + lo * NK + k0 + 8 * hi);
      h8 v1 = *(const h8*)(vp + lo * NK + k0 + 16 + 8 * hi);
      vb = SHUF16(v0, v1);
    }
    acc = WMMA(SHUF16(p0, p1), vb, acc);
  }
  if (lo < 8) {                                      // a_t[b][qp][d], K-minor
#pragma unroll
    for (int r = 0; r < 8; ++r) {
      int qp = qtile * 16 + r + 8 * hi;
      if (qp < NQ)
        a_t[((size_t)b * NQ + qp) * DV + head * 8 + lo] = (_Float16)(acc[r] / l_run[r]);
    }
  }
}

// ---------------------------------------------------------------------------
// Output projection: out[b,co,qp] += sum_d w_o[co,d] * a_t[b,qp,d].
__global__ __launch_bounds__(32)
void outproj_wmma_kernel(const _Float16* __restrict__ a_t,
                         const _Float16* __restrict__ woh, // [128][64]
                         float* __restrict__ out) {
  const int lane = threadIdx.x;
  const int hi = lane >> 4, lo = lane & 15;
  const int ntile = blockIdx.x, mtile = blockIdx.y, b = blockIdx.z;
  const int qp = ntile * 16 + lo;
  const _Float16* wrow = woh + (size_t)(mtile * 16 + lo) * DV;

  v8f acc = {};
#pragma unroll
  for (int kc = 0; kc < DV; kc += 32) {
    h8 a0 = *(const h8*)(wrow + kc + 8 * hi);
    h8 a1 = *(const h8*)(wrow + kc + 16 + 8 * hi);
    v16h bv = {};
    if (qp < NQ) {
      h8 b0 = *(const h8*)(a_t + ((size_t)b * NQ + qp) * DV + kc + 8 * hi);
      h8 b1 = *(const h8*)(a_t + ((size_t)b * NQ + qp) * DV + kc + 16 + 8 * hi);
      bv = SHUF16(b0, b1);
    }
    acc = WMMA(SHUF16(a0, a1), bv, acc);
  }
  if (qp < NQ) {
#pragma unroll
    for (int r = 0; r < 8; ++r) {
      int co = mtile * 16 + r + 8 * hi;
      out[((size_t)(b * CO + co)) * NQ + qp] += acc[r];
    }
  }
}

// ---------------------------------------------------------------------------
extern "C" void kernel_launch(void* const* d_in, const int* in_sizes, int n_in,
                              void* d_out, int out_size, void* d_ws, size_t ws_size,
                              hipStream_t stream) {
  const float* x     = (const float*)d_in[0];
  const float* gamma = (const float*)d_in[1];
  const float* beta  = (const float*)d_in[2];
  const float* rmean = (const float*)d_in[3];
  const float* rvar  = (const float*)d_in[4];
  const float* w_sc  = (const float*)d_in[5];   // [256][128][3][3]
  const float* w_q   = (const float*)d_in[6];   // [256][64][3][3]
  const float* w_k   = (const float*)d_in[7];   // [64][256]
  const float* w_v   = (const float*)d_in[8];   // [64][256]
  const float* w_o   = (const float*)d_in[9];   // [128][64]
  float* out = (float*)d_out;                   // [4][128][65][65]

  char* ws = (char*)d_ws;
  size_t off = 0;
  auto take = [&](size_t bytes) { void* p = ws + off; off = (off + bytes + 255) & ~(size_t)255; return p; };
  _Float16* xr_t  = (_Float16*)take((size_t)BB * HWIN * CIN * 2);   // 2 MiB
  _Float16* wscT  = (_Float16*)take((size_t)9 * CO * CIN * 2);
  _Float16* wqT   = (_Float16*)take((size_t)9 * DK * CIN * 2);
  _Float16* wk_h  = (_Float16*)take((size_t)DK * CIN * 2);
  _Float16* wv_h  = (_Float16*)take((size_t)DV * CIN * 2);
  _Float16* wo_h  = (_Float16*)take((size_t)CO * DV * 2);
  float*    q_acc = (float*)take((size_t)BB * DK * NQ * 4);         // 4.3 MiB
  _Float16* q_t   = (_Float16*)take((size_t)BB * NQ * DK * 2);      // 2.2 MiB
  _Float16* k_t   = (_Float16*)take((size_t)BB * HWIN * DK * 2);
  _Float16* v_h   = (_Float16*)take((size_t)BB * DV * NK * 2);
  _Float16* a_t   = (_Float16*)take((size_t)BB * NQ * DV * 2);      // 2.2 MiB
  (void)ws_size; (void)n_in; (void)in_sizes; (void)out_size;

  const long n_out = (long)BB * CO * NQ;        // 2,163,200
  const long n_q   = (long)BB * DK * NQ;        // 1,081,600

  zero2_kernel<<<(unsigned)((n_out + 255) / 256), 256, 0, stream>>>(out, n_out, q_acc, n_q);
  bn_relu_pack_kernel<<<(BB * HWIN * CIN) / 256, 256, 0, stream>>>(x, gamma, beta, rmean, rvar, xr_t);

  pack_wT_kernel<<<(9 * CO * CIN + 255) / 256, 256, 0, stream>>>(w_sc, wscT, CO);
  pack_wT_kernel<<<(9 * DK * CIN + 255) / 256, 256, 0, stream>>>(w_q, wqT, DK);
  cvt_f16_kernel<<<(DK * CIN + 255) / 256, 256, 0, stream>>>(w_k, wk_h, DK * CIN);
  cvt_f16_kernel<<<(DV * CIN + 255) / 256, 256, 0, stream>>>(w_v, wv_h, DV * CIN);
  cvt_f16_kernel<<<(CO * DV + 255) / 256, 256, 0, stream>>>(w_o, wo_h, CO * DV);

  // ConvTranspose GEMMs: block = (Cout/16) waves sharing one staged B tile
  convt_wmma_kernel<<<dim3(HWIN / 16, 1, BB), (CO / 16) * 32, 0, stream>>>(xr_t, wscT, out, CO);
  convt_wmma_kernel<<<dim3(HWIN / 16, 1, BB), (DK / 16) * 32, 0, stream>>>(xr_t, wqT, q_acc, DK);

  pack_q_kernel<<<(BB * NQ * DK + 255) / 256, 256, 0, stream>>>(q_acc, q_t);

  kv_wmma_kernel<<<dim3(NK / 16, 1, BB), 128, 0, stream>>>(xr_t, wk_h, k_t, 1);
  kv_wmma_kernel<<<dim3(NK / 16, 1, BB), 128, 0, stream>>>(xr_t, wv_h, v_h, 0);

  attn_flash_kernel<<<dim3(QTILES, HEADS, BB), 32, 0, stream>>>(q_t, k_t, v_h, a_t);
  outproj_wmma_kernel<<<dim3(QTILES, CO / 16, BB), 32, 0, stream>>>(a_t, wo_h, out);
}
